// DecoderLayer_13872744366274
// MI455X (gfx1250) — matrix operand order repaired and checked by reading, not compiled, not run on hardware
//
#include <hip/hip_runtime.h>
#include <hip/hip_bf16.h>

#define D_MODEL 1024
#define N_HEADS 16
#define HEAD    64
#define D_FF    4096
#define TSEQ    4096
#define LN_EPS  1e-5f

typedef __bf16 bf16;
typedef __attribute__((ext_vector_type(8)))  bf16  v8bf;
typedef __attribute__((ext_vector_type(16))) bf16  v16bf;
typedef __attribute__((ext_vector_type(8)))  float v8f;
typedef __attribute__((ext_vector_type(4)))  unsigned int u32x4;
typedef __attribute__((ext_vector_type(8)))  int i32x8;
typedef __attribute__((ext_vector_type(4)))  int i32x4;

static __device__ __forceinline__ v16bf mk16(v8bf lo, v8bf hi) {
  return __builtin_shufflevector(lo, hi, 0,1,2,3,4,5,6,7,8,9,10,11,12,13,14,15);
}
static __device__ __forceinline__ v8bf ld8(const bf16* p) { return *(const v8bf*)p; }
static __device__ __forceinline__ v8f zero8() {
  v8f z = {0.f,0.f,0.f,0.f,0.f,0.f,0.f,0.f};
  return z;
}
static __device__ __forceinline__ v8f wmma_bf16(v16bf a, v16bf b, v8f c) {
  // D = A(16x32) * B(32x16) + C, f32 accumulate
  return __builtin_amdgcn_wmma_f32_16x16x32_bf16(false, a, false, b, (short)0, c, false, false);
}

#define LDS_STRIDE 40   // padded LDS row stride in bf16 (64B row + 16B pad)

// ---------------------------------------------------------------------------
// TDM: DMA a 128-row x 32-col bf16 tile (row stride ld_elems) from global
// memory into LDS at byte offset lds_byte_off, writing padded rows of
// LDS_STRIDE bf16 (pad_interval = 64B, pad_amount = 16B). Tracked by TENSORcnt.
// ---------------------------------------------------------------------------
static __device__ __forceinline__ void tdm_load_tile_128x32(
    unsigned lds_byte_off, const bf16* gsrc, int ld_elems) {
  const unsigned long long ga = (unsigned long long)(size_t)gsrc;
  u32x4 g0;
  g0[0] = 1u;                                          // count=1 (valid user D#)
  g0[1] = lds_byte_off;                                // lds_addr
  g0[2] = (unsigned)ga;                                // global_addr[31:0]
  g0[3] = ((unsigned)(ga >> 32) & 0x01ffffffu)         // global_addr[56:32]
        | (2u << 30);                                  // type = 2 ("image")
  i32x8 g1;
  g1[0] = (1 << 16)                                    // data_size = 2 bytes
        | (1 << 20)                                    // pad_enable
        | (3 << 22)                                    // pad_interval: 16 DW = 64B
        | (3 << 25);                                   // pad_amount: 4 DW = 16B
  g1[1] = (int)(32u << 16);                            // tensor_dim0 = 32
  g1[2] = (int)(128u << 16);                           // tensor_dim1 = 128
  g1[3] = (int)(32u << 16);                            // tile_dim0 = 32
  g1[4] = 128;                                         // tile_dim1 = 128, tile_dim2 = 0
  g1[5] = ld_elems;                                    // tensor_dim0_stride
  g1[6] = 0;
  g1[7] = 0;
  const i32x4 gz4 = {0, 0, 0, 0};                      // groups 2/3: 2-D tile
  const i32x8 gz8 = {0, 0, 0, 0, 0, 0, 0, 0};
  __builtin_amdgcn_tensor_load_to_lds(g0, g1, gz4, gz4, gz8, 0);
}

// ---------------------------------------------------------------------------
// LayerNorm: fp32 [T, D_MODEL] -> bf16 [T, D_MODEL]
// ---------------------------------------------------------------------------
__global__ __launch_bounds__(256) void ln_bf16_kernel(
    const float* __restrict__ x, const float* __restrict__ g,
    const float* __restrict__ b, bf16* __restrict__ out) {
  const int row = blockIdx.x, tid = threadIdx.x;
  const float4 v = ((const float4*)(x + (size_t)row * D_MODEL))[tid];
  __shared__ float red[256];
  red[tid] = v.x + v.y + v.z + v.w;
  __syncthreads();
  #pragma unroll
  for (int s = 128; s > 0; s >>= 1) {
    if (tid < s) red[tid] += red[tid + s];
    __syncthreads();
  }
  const float mu = red[0] * (1.0f / D_MODEL);
  __syncthreads();
  const float dx = v.x - mu, dy = v.y - mu, dz = v.z - mu, dw = v.w - mu;
  red[tid] = dx*dx + dy*dy + dz*dz + dw*dw;
  __syncthreads();
  #pragma unroll
  for (int s = 128; s > 0; s >>= 1) {
    if (tid < s) red[tid] += red[tid + s];
    __syncthreads();
  }
  const float rstd = rsqrtf(red[0] * (1.0f / D_MODEL) + LN_EPS);
  const float4 gv = ((const float4*)g)[tid];
  const float4 bv = ((const float4*)b)[tid];
  bf16* o = out + (size_t)row * D_MODEL + tid * 4;
  o[0] = (bf16)(dx * rstd * gv.x + bv.x);
  o[1] = (bf16)(dy * rstd * gv.y + bv.y);
  o[2] = (bf16)(dz * rstd * gv.z + bv.z);
  o[3] = (bf16)(dw * rstd * gv.w + bv.w);
}

// ---------------------------------------------------------------------------
// Weight convert + transpose: fp32 W[K][N] -> bf16 Wt[N][K]
// ---------------------------------------------------------------------------
__global__ __launch_bounds__(256) void convert_wt_kernel(
    const float* __restrict__ W, bf16* __restrict__ Wt, int K, int N) {
  const int idx = blockIdx.x * 256 + threadIdx.x;
  if (idx >= K * N) return;
  const int k = idx / N, n = idx - k * N;
  Wt[(size_t)n * K + k] = (bf16)W[idx];
}

// ---------------------------------------------------------------------------
// bf16 WMMA GEMM: C[M,N] = A[M,K] * Bt[N,K]^T + bias, templated epilogue.
// Block 256 thr = 8 waves; block tile 128x128; wave tile 32x64; K-step 32.
// A/B tiles DMA'd into double-buffered LDS by the Tensor Data Mover (wave 0
// issues, TENSORcnt-waited, barrier-released); waves only do ds_load_b128
// fragment reads + WMMA.
// Dynamic LDS layout (byte offsets): As0 @0, As1 @10240, Bs0 @20480, Bs1 @30720.
// ---------------------------------------------------------------------------
#define EPI_QKV  0
#define EPI_RES  1
#define EPI_GELU 2
#define TILE_BYTES (128 * LDS_STRIDE * 2)   // 10240
#define GEMM_LDS_BYTES (4 * TILE_BYTES)     // 40960

template<int EPI>
__global__ __launch_bounds__(256) void gemm_bf16_kernel(
    const bf16* __restrict__ A, const bf16* __restrict__ Bt,
    const float* __restrict__ bias, int M, int N, int K,
    float* __restrict__ outF, const float* __restrict__ resid,
    bf16* __restrict__ outB,
    bf16* __restrict__ qout, bf16* __restrict__ kout, bf16* __restrict__ vtout) {
  extern __shared__ bf16 smem[];   // only LDS object in this kernel -> offset 0
  const int tid   = threadIdx.x;
  const int lane  = tid & 31, lhalf = lane & 15, hsel = lane >> 4;
  const int wid   = tid >> 5, wm = wid & 3, wn = wid >> 2;
  const int bm    = blockIdx.y * 128, bn = blockIdx.x * 128;
  const bf16* Arow = A  + (size_t)bm * K;
  const bf16* Brow = Bt + (size_t)bn * K;

  v8f acc[2][4];
  #pragma unroll
  for (int i = 0; i < 2; i++)
    #pragma unroll
    for (int j = 0; j < 4; j++) acc[i][j] = zero8();

  const int nk = K >> 5;
  if (wid == 0) {
    tdm_load_tile_128x32(0u,              Arow, K);
    tdm_load_tile_128x32(2u * TILE_BYTES, Brow, K);
  }
  for (int t = 0; t < nk; ++t) {
    const int cur = t & 1;
    if (wid == 0) {
      if (t + 1 < nk) {   // prefetch next tile into the other buffer
        tdm_load_tile_128x32((unsigned)((cur ^ 1) * TILE_BYTES),
                             Arow + (size_t)(t + 1) * 32, K);
        tdm_load_tile_128x32((unsigned)(2 * TILE_BYTES + (cur ^ 1) * TILE_BYTES),
                             Brow + (size_t)(t + 1) * 32, K);
        __builtin_amdgcn_s_wait_tensorcnt(2);   // current tile's 2 loads done
      } else {
        __builtin_amdgcn_s_wait_tensorcnt(0);
      }
    }
    __syncthreads();   // release compute waves once tile `t` is resident

    const bf16* As = smem + cur * (128 * LDS_STRIDE);
    const bf16* Bs = smem + (2 + cur) * (128 * LDS_STRIDE);
    v16bf af[2], bfr[4];
    #pragma unroll
    for (int fm = 0; fm < 2; fm++) {
      const bf16* p = &As[(wm * 32 + fm * 16 + lhalf) * LDS_STRIDE];
      af[fm] = mk16(*(const v8bf*)(p + hsel * 8), *(const v8bf*)(p + 16 + hsel * 8));
    }
    #pragma unroll
    for (int fn = 0; fn < 4; fn++) {
      const bf16* p = &Bs[(wn * 64 + fn * 16 + lhalf) * LDS_STRIDE];
      bfr[fn] = mk16(*(const v8bf*)(p + hsel * 8), *(const v8bf*)(p + 16 + hsel * 8));
    }
    #pragma unroll
    for (int fm = 0; fm < 2; fm++)
      #pragma unroll
      for (int fn = 0; fn < 4; fn++)
        acc[fm][fn] = wmma_bf16(af[fm], bfr[fn], acc[fm][fn]);

    __syncthreads();   // compute done before TDM may overwrite this buffer
  }

  // Epilogue. C layout: VGPR r -> row (r + 8*hsel), col = lhalf within frag.
  #pragma unroll
  for (int fm = 0; fm < 2; fm++) {
    #pragma unroll
    for (int fn = 0; fn < 4; fn++) {
      const int col = bn + wn * 64 + fn * 16 + lhalf;
      const float bv = bias[col];
      #pragma unroll
      for (int r = 0; r < 8; r++) {
        const int row = bm + wm * 32 + fm * 16 + r + hsel * 8;
        const float v = acc[fm][fn][r] + bv;
        if (EPI == EPI_RES) {
          const size_t o = (size_t)row * N + col;
          outF[o] = v + resid[o];
        } else if (EPI == EPI_GELU) {
          outB[(size_t)row * N + col] =
              (bf16)(0.5f * v * (1.0f + erff(v * 0.70710678118654752f)));
        } else {  // EPI_QKV: split into q [h][T][64], k [h][T][64], vT [h][64][T]
          if (col < D_MODEL) {
            const int h = col >> 6, d = col & 63;
            qout[((size_t)h * TSEQ + row) * HEAD + d] = (bf16)v;
          } else if (col < 2 * D_MODEL) {
            const int nn = col - D_MODEL, h = nn >> 6, d = nn & 63;
            kout[((size_t)h * TSEQ + row) * HEAD + d] = (bf16)v;
          } else {
            const int nn = col - 2 * D_MODEL, h = nn >> 6, d = nn & 63;
            vtout[((size_t)h * HEAD + d) * TSEQ + row] = (bf16)v;
          }
        }
      }
    }
  }
}

// ---------------------------------------------------------------------------
// Flash attention: one wave per (head, 16-query tile). Causal + key mask.
// Q,K: [h][T][64] bf16;  Vt: [h][64][T] bf16;  Y: [T, D_MODEL] bf16.
// ---------------------------------------------------------------------------
__global__ __launch_bounds__(32) void attn_kernel(
    const bf16* __restrict__ Q, const bf16* __restrict__ Kb,
    const bf16* __restrict__ Vt, const int* __restrict__ amask,
    bf16* __restrict__ Y) {
  __shared__ __align__(16) bf16 Sl[16 * LDS_STRIDE];
  const int lane = threadIdx.x, lhalf = lane & 15, hsel = lane >> 4;
  const int head = blockIdx.y;
  const int q0   = blockIdx.x * 16;
  const bf16* qp = Q  + ((size_t)head * TSEQ + q0) * HEAD;
  const bf16* kp = Kb + (size_t)head * TSEQ * HEAD;
  const bf16* vp = Vt + (size_t)head * HEAD * TSEQ;

  v16bf qf[2];
  #pragma unroll
  for (int f = 0; f < 2; f++) {
    const bf16* r = qp + lhalf * HEAD + f * 32;
    qf[f] = mk16(ld8(r + hsel * 8), ld8(r + 16 + hsel * 8));
  }

  v8f o[4];
  #pragma unroll
  for (int d = 0; d < 4; d++) o[d] = zero8();
  float mrow[8], lrow[8];
  #pragma unroll
  for (int r = 0; r < 8; r++) { mrow[r] = -1e30f; lrow[r] = 0.f; }

  const float scale = 0.125f;  // 1/sqrt(64)
  const int kend = q0 + 16;
  for (int k0 = 0; k0 < kend; k0 += 32) {
    // S = Q * K^T for 32 keys as two 16-wide N-tiles
    v8f s0 = zero8(), s1 = zero8();
    #pragma unroll
    for (int kk = 0; kk < 2; kk++) {
      const bf16* kr0 = kp + (size_t)(k0 + lhalf)      * HEAD + kk * 32 + hsel * 8;
      const bf16* kr1 = kp + (size_t)(k0 + 16 + lhalf) * HEAD + kk * 32 + hsel * 8;
      s0 = wmma_bf16(qf[kk], mk16(ld8(kr0), ld8(kr0 + 16)), s0);
      s1 = wmma_bf16(qf[kk], mk16(ld8(kr1), ld8(kr1 + 16)), s1);
    }
    const int key0 = k0 + lhalf, key1 = k0 + 16 + lhalf;
    const bool am0 = amask[key0] != 0, am1 = amask[key1] != 0;

    // online softmax; row stats across the 16-lane half-wave groups
    #pragma unroll
    for (int r = 0; r < 8; r++) {
      const int qi = q0 + r + hsel * 8;
      float v0 = (am0 && key0 <= qi) ? s0[r] * scale : -1e30f;
      float v1 = (am1 && key1 <= qi) ? s1[r] * scale : -1e30f;
      float bm = fmaxf(v0, v1);
      #pragma unroll
      for (int m = 1; m < 16; m <<= 1) bm = fmaxf(bm, __shfl_xor(bm, m, 32));
      const float nm   = fmaxf(mrow[r], bm);
      const float corr = __expf(mrow[r] - nm);
      const float p0   = __expf(v0 - nm);
      const float p1   = __expf(v1 - nm);
      float rs = p0 + p1;
      #pragma unroll
      for (int m = 1; m < 16; m <<= 1) rs += __shfl_xor(rs, m, 32);
      lrow[r] = lrow[r] * corr + rs;
      mrow[r] = nm;
      o[0][r] *= corr; o[1][r] *= corr; o[2][r] *= corr; o[3][r] *= corr;
      Sl[(r + hsel * 8) * LDS_STRIDE + lhalf]      = (bf16)p0;
      Sl[(r + hsel * 8) * LDS_STRIDE + 16 + lhalf] = (bf16)p1;
    }
    __syncthreads();  // C-layout -> A-layout relayout through LDS

    const bf16* pr = &Sl[lhalf * LDS_STRIDE];
    const v16bf pf = mk16(*(const v8bf*)(pr + hsel * 8),
                          *(const v8bf*)(pr + 16 + hsel * 8));
    #pragma unroll
    for (int d = 0; d < 4; d++) {
      const bf16* vr = vp + (size_t)(d * 16 + lhalf) * TSEQ + k0 + hsel * 8;
      o[d] = wmma_bf16(pf, mk16(ld8(vr), ld8(vr + 16)), o[d]);
    }
    __syncthreads();
  }

  #pragma unroll
  for (int d = 0; d < 4; d++) {
    const int col = head * HEAD + d * 16 + lhalf;
    #pragma unroll
    for (int r = 0; r < 8; r++) {
      const int row = q0 + r + hsel * 8;
      const float inv = lrow[r] > 0.f ? 1.0f / lrow[r] : 0.f;
      Y[(size_t)row * D_MODEL + col] = (bf16)(o[d][r] * inv);
    }
  }
}

// ---------------------------------------------------------------------------
extern "C" void kernel_launch(void* const* d_in, const int* in_sizes, int n_in,
                              void* d_out, int out_size, void* d_ws, size_t ws_size,
                              hipStream_t stream) {
  const float* x     = (const float*)d_in[0];
  const int*   amask = (const int*)d_in[1];
  const float* ln1_g = (const float*)d_in[2];
  const float* ln1_b = (const float*)d_in[3];
  const float* W_qkv = (const float*)d_in[4];
  const float* b_qkv = (const float*)d_in[5];
  const float* W_o   = (const float*)d_in[6];
  const float* b_o   = (const float*)d_in[7];
  const float* ln2_g = (const float*)d_in[8];
  const float* ln2_b = (const float*)d_in[9];
  const float* W1    = (const float*)d_in[10];
  const float* b1    = (const float*)d_in[11];
  const float* W2    = (const float*)d_in[12];
  const float* b2    = (const float*)d_in[13];
  float* out = (float*)d_out;
  (void)in_sizes; (void)n_in; (void)out_size; (void)ws_size;

  char* ws = (char*)d_ws;
  size_t off = 0;
  auto take = [&](size_t bytes) -> void* {
    void* p = ws + off;
    off += (bytes + 255) & ~(size_t)255;
    return p;
  };
  // hbf is reused for: ln1 output -> attention output y -> ln2 output
  bf16* hbf   = (bf16*)take((size_t)TSEQ * D_MODEL * 2);   // 8 MB
  bf16* qbf   = (bf16*)take((size_t)TSEQ * D_MODEL * 2);   // 8 MB
  bf16* kbf   = (bf16*)take((size_t)TSEQ * D_MODEL * 2);   // 8 MB
  bf16* vtbf  = (bf16*)take((size_t)TSEQ * D_MODEL * 2);   // 8 MB
  bf16* h2    = (bf16*)take((size_t)TSEQ * D_FF * 2);      // 32 MB
  bf16* Wqkvt = (bf16*)take((size_t)D_MODEL * 3 * D_MODEL * 2); // 6 MB
  bf16* Wot   = (bf16*)take((size_t)D_MODEL * D_MODEL * 2);     // 2 MB
  bf16* W1t   = (bf16*)take((size_t)D_MODEL * D_FF * 2);        // 8 MB
  bf16* W2t   = (bf16*)take((size_t)D_FF * D_MODEL * 2);        // 8 MB

  // 1) weight convert + transpose to bf16 [N][K]
  convert_wt_kernel<<<(D_MODEL * 3 * D_MODEL + 255) / 256, 256, 0, stream>>>(
      W_qkv, Wqkvt, D_MODEL, 3 * D_MODEL);
  convert_wt_kernel<<<(D_MODEL * D_MODEL + 255) / 256, 256, 0, stream>>>(
      W_o, Wot, D_MODEL, D_MODEL);
  convert_wt_kernel<<<(D_MODEL * D_FF + 255) / 256, 256, 0, stream>>>(
      W1, W1t, D_MODEL, D_FF);
  convert_wt_kernel<<<(D_FF * D_MODEL + 255) / 256, 256, 0, stream>>>(
      W2, W2t, D_FF, D_MODEL);

  // 2) LN1
  ln_bf16_kernel<<<TSEQ, 256, 0, stream>>>(x, ln1_g, ln1_b, hbf);

  // 3) QKV projection (split epilogue)
  gemm_bf16_kernel<EPI_QKV>
      <<<dim3(3 * D_MODEL / 128, TSEQ / 128), 256, GEMM_LDS_BYTES, stream>>>(
      hbf, Wqkvt, b_qkv, TSEQ, 3 * D_MODEL, D_MODEL,
      nullptr, nullptr, nullptr, qbf, kbf, vtbf);

  // 4) flash attention -> y (reuse hbf)
  attn_kernel<<<dim3(TSEQ / 16, N_HEADS), 32, 0, stream>>>(qbf, kbf, vtbf, amask, hbf);

  // 5) output projection + residual: out = x + y@Wo + b_o
  gemm_bf16_kernel<EPI_RES>
      <<<dim3(D_MODEL / 128, TSEQ / 128), 256, GEMM_LDS_BYTES, stream>>>(
      hbf, Wot, b_o, TSEQ, D_MODEL, D_MODEL,
      out, x, nullptr, nullptr, nullptr, nullptr);

  // 6) LN2 (reads out, writes hbf)
  ln_bf16_kernel<<<TSEQ, 256, 0, stream>>>(out, ln2_g, ln2_b, hbf);

  // 7) MLP up + exact GELU
  gemm_bf16_kernel<EPI_GELU>
      <<<dim3(D_FF / 128, TSEQ / 128), 256, GEMM_LDS_BYTES, stream>>>(
      hbf, W1t, b1, TSEQ, D_FF, D_MODEL,
      nullptr, nullptr, h2, nullptr, nullptr, nullptr);

  // 8) MLP down + residual (in-place read/write of out by same lane is safe)
  gemm_bf16_kernel<EPI_RES>
      <<<dim3(D_MODEL / 128, TSEQ / 128), 256, GEMM_LDS_BYTES, stream>>>(
      h2, W2t, b2, TSEQ, D_MODEL, D_FF,
      out, out, nullptr, nullptr, nullptr, nullptr);
}
